// GraphSAGE_26285199852117
// MI455X (gfx1250) — matrix operand order repaired and checked by reading, not compiled, not run on hardware
//
#include <hip/hip_runtime.h>
#include <hip/hip_bf16.h>

#define N_NODES 50000
#define N_EDGES 800000
#define IN_DIM  64
#define HID     128
#define BN_EPS  1e-5f

typedef __attribute__((ext_vector_type(16))) _Float16 v16h;
typedef __attribute__((ext_vector_type(8)))  _Float16 v8h;
typedef __attribute__((ext_vector_type(8)))  float    v8f;
typedef __attribute__((ext_vector_type(4)))  float    v4f;

// ---------------- utility kernels ----------------

__global__ void zero_f32(float* __restrict__ p, int n) {
    int i = blockIdx.x * blockDim.x + threadIdx.x;
    if (i < n) p[i] = 0.0f;
}

__global__ void cvt_f32_to_f16(const float* __restrict__ s, _Float16* __restrict__ d, int n) {
    int i = blockIdx.x * blockDim.x + threadIdx.x;
    if (i < n) d[i] = (_Float16)s[i];
}

// ---------------- graph aggregation (mean over in-neighbors) ----------------
// 16 lanes per edge, 8 contiguous f16 features per lane:
//   - one b128 gather of h[src] per lane (16B aligned: row stride 256B)
//   - 8 scalar fp32 atomic adds into sum[dst]
//   - lane 0 of each edge also bumps the degree counter (fused count pass)

__global__ void scatter_kernel(const int* __restrict__ src, const int* __restrict__ dst,
                               const _Float16* __restrict__ h,
                               float* __restrict__ sum, float* __restrict__ cnt) {
    int i = blockIdx.x * blockDim.x + threadIdx.x;     // E*16 threads
    if (i < N_EDGES * 16) {
        int e   = i >> 4;
        int sub = i & 15;
        int f0  = sub * 8;
        int s   = src[e];
        int d   = dst[e];
        v8h val = *(const v8h*)(h + (size_t)s * HID + f0);
        float* out = sum + (size_t)d * HID + f0;
#pragma unroll
        for (int j = 0; j < 8; ++j) atomicAdd(&out[j], (float)val[j]);
        if (sub == 0) atomicAdd(&cnt[d], 1.0f);
    }
}

// scatter-mean -> f16, 8 features per thread (two float4 loads, one 16B store)
__global__ void aggr_kernel(const float* __restrict__ sum, const float* __restrict__ cnt,
                            _Float16* __restrict__ out) {
    int i = blockIdx.x * blockDim.x + threadIdx.x;     // N*HID/8 threads
    if (i < N_NODES * HID / 8) {
        int base  = i * 8;
        float inv = 1.0f / fmaxf(cnt[base >> 7], 1.0f);
        v4f a = *(const v4f*)(sum + base);
        v4f b = *(const v4f*)(sum + base + 4);
        v8h o;
#pragma unroll
        for (int j = 0; j < 4; ++j) { o[j] = (_Float16)(a[j] * inv); o[j + 4] = (_Float16)(b[j] * inv); }
        *(v8h*)(out + base) = o;
    }
}

// ---------------- fused WMMA GEMM: C = A1 @ W1^T [+ A2 @ W2^T] + bias ----------------
// A*: [M_total, K] f16 row-major.  W*: [128, K] f16 row-major (torch-style weight).
// One block = 16 rows x 128 cols; wave w owns column tile w (wave32, EXEC all-ones).
// Lane layouts per CDNA5 ISA 7.12.2 (16-bit A 16x32, 16-bit B 32x16, f32 C/D).

__global__ void wmma_gemm(const _Float16* __restrict__ A1, const _Float16* __restrict__ W1, int K1,
                          const _Float16* __restrict__ A2, const _Float16* __restrict__ W2, int K2,
                          const float* __restrict__ bias,
                          float* __restrict__ Cf, _Float16* __restrict__ Ch) {
    const int lane = threadIdx.x & 31;
    const int wave = threadIdx.x >> 5;      // 0..7 -> output column tile
    const int l16  = lane & 15;
    const int hi   = lane >> 4;             // lane half: selects K sub-range
    const int ocol = wave * 16 + l16;       // output column (N index of B/C)
    const int mrow = blockIdx.x * 16 + l16; // A row (M index)

    v8f acc;
    float bv = bias ? bias[ocol] : 0.0f;
#pragma unroll
    for (int r = 0; r < 8; ++r) acc[r] = bv;

    {
        const _Float16* arow = A1 + (size_t)mrow * K1;
        const _Float16* wrow = W1 + (size_t)ocol * K1;
        for (int kb = 0; kb < K1; kb += 32) {
            v16h a, b;
            const int ka = kb + hi * 8;     // A: lanes>=16 cover K+8..15 / K+24..31
#pragma unroll
            for (int j = 0; j < 8; ++j) { a[j] = arow[ka + j]; a[j + 8] = arow[ka + 16 + j]; }
#pragma unroll
            for (int j = 0; j < 8; ++j) { b[j] = wrow[ka + j]; b[j + 8] = wrow[ka + 16 + j]; }
            acc = __builtin_amdgcn_wmma_f32_16x16x32_f16(false, a, false, b,
                                                         (short)0, acc, false, false);
        }
    }
    if (A2) {
        const _Float16* arow = A2 + (size_t)mrow * K2;
        const _Float16* wrow = W2 + (size_t)ocol * K2;
        for (int kb = 0; kb < K2; kb += 32) {
            v16h a, b;
            const int ka = kb + hi * 8;
#pragma unroll
            for (int j = 0; j < 8; ++j) { a[j] = arow[ka + j]; a[j + 8] = arow[ka + 16 + j]; }
#pragma unroll
            for (int j = 0; j < 8; ++j) { b[j] = wrow[ka + j]; b[j + 8] = wrow[ka + 16 + j]; }
            acc = __builtin_amdgcn_wmma_f32_16x16x32_f16(false, a, false, b,
                                                         (short)0, acc, false, false);
        }
    }
    // C/D layout: VGPR r -> M = r + hi*8, N = l16
#pragma unroll
    for (int r = 0; r < 8; ++r) {
        int m = blockIdx.x * 16 + r + hi * 8;
        if (Cf) Cf[(size_t)m * HID + ocol] = acc[r];
        if (Ch) Ch[(size_t)m * HID + ocol] = (_Float16)acc[r];
    }
}

// ---------------- BatchNorm (training-mode batch stats) + ReLU ----------------

__global__ void bn_stats(const float* __restrict__ h, float* __restrict__ stats) {
    int f = threadIdx.x;              // blockDim = 128 channels
    int row0 = blockIdx.x * 256;
    float s1 = 0.0f, s2 = 0.0f;
    for (int r = 0; r < 256; ++r) {
        int node = row0 + r;
        if (node < N_NODES) {
            float v = h[(size_t)node * HID + f];
            s1 += v; s2 += v * v;
        }
    }
    atomicAdd(&stats[f], s1);
    atomicAdd(&stats[HID + f], s2);
}

__global__ void bn_finalize(const float* __restrict__ stats, const float* __restrict__ gamma,
                            const float* __restrict__ beta, float* __restrict__ ss) {
    int f = threadIdx.x;              // 128 threads
    float mu  = stats[f] / (float)N_NODES;
    float var = stats[HID + f] / (float)N_NODES - mu * mu;   // biased var
    float sc  = gamma[f] * rsqrtf(var + BN_EPS);
    ss[f] = sc;
    ss[HID + f] = beta[f] - mu * sc;
}

__global__ void bn_apply(const float* __restrict__ h, const float* __restrict__ ss,
                         _Float16* __restrict__ out) {
    int i = blockIdx.x * blockDim.x + threadIdx.x;     // N*HID/8 threads
    if (i < N_NODES * HID / 8) {
        int base = i * 8;
        int f0   = base & (HID - 1);
        v4f a = *(const v4f*)(h + base);
        v4f b = *(const v4f*)(h + base + 4);
        v8h o;
#pragma unroll
        for (int j = 0; j < 4; ++j) {
            o[j]     = (_Float16)fmaxf(a[j] * ss[f0 + j]     + ss[HID + f0 + j],     0.0f);
            o[j + 4] = (_Float16)fmaxf(b[j] * ss[f0 + j + 4] + ss[HID + f0 + j + 4], 0.0f);
        }
        *(v8h*)(out + base) = o;
    }
}

// ---------------- classifier (OUT=2) + softmax ----------------

__global__ void classifier(const _Float16* __restrict__ h, const float* __restrict__ Wc,
                           const float* __restrict__ bc, float* __restrict__ out) {
    int n = blockIdx.x * blockDim.x + threadIdx.x;
    if (n < N_NODES) {
        float a0 = bc[0], a1 = bc[1];
        const _Float16* row = h + (size_t)n * HID;
        for (int k = 0; k < HID; ++k) {
            float v = (float)row[k];
            a0 += v * Wc[k];
            a1 += v * Wc[HID + k];
        }
        float m = fmaxf(a0, a1);
        float e0 = __expf(a0 - m), e1 = __expf(a1 - m);
        float inv = 1.0f / (e0 + e1);
        out[n * 2 + 0] = e0 * inv;
        out[n * 2 + 1] = e1 * inv;
    }
}

// ---------------- host orchestration ----------------

extern "C" void kernel_launch(void* const* d_in, const int* in_sizes, int n_in,
                              void* d_out, int out_size, void* d_ws, size_t ws_size,
                              hipStream_t stream) {
    (void)in_sizes; (void)n_in; (void)out_size; (void)ws_size;

    const float* x     = (const float*)d_in[0];
    const int*   ei    = (const int*)d_in[1];
    const int*   src   = ei;                 // edge_index[0]
    const int*   dst   = ei + N_EDGES;       // edge_index[1]
    const float* W_emb = (const float*)d_in[2];
    const float* b_emb = (const float*)d_in[3];
    const float* Wc    = (const float*)d_in[4];
    const float* bc    = (const float*)d_in[5];
    const float* Wl[3] = {(const float*)d_in[6],  (const float*)d_in[11], (const float*)d_in[16]};
    const float* bl[3] = {(const float*)d_in[7],  (const float*)d_in[12], (const float*)d_in[17]};
    const float* Wr[3] = {(const float*)d_in[8],  (const float*)d_in[13], (const float*)d_in[18]};
    const float* g[3]  = {(const float*)d_in[9],  (const float*)d_in[14], (const float*)d_in[19]};
    const float* be[3] = {(const float*)d_in[10], (const float*)d_in[15], (const float*)d_in[20]};

    // workspace carve-up (~71 MB)
    char* ws = (char*)d_ws;
    size_t off = 0;
    auto carve = [&](size_t bytes) -> char* {
        char* p = ws + off;
        off = (off + bytes + 255) & ~(size_t)255;
        return p;
    };
    _Float16* xf16   = (_Float16*)carve((size_t)N_NODES * IN_DIM * sizeof(_Float16));
    _Float16* we16   = (_Float16*)carve((size_t)HID * IN_DIM * sizeof(_Float16));
    _Float16* wl16[3];
    _Float16* wr16[3];
    for (int i = 0; i < 3; ++i) {
        wl16[i] = (_Float16*)carve((size_t)HID * HID * sizeof(_Float16));
        wr16[i] = (_Float16*)carve((size_t)HID * HID * sizeof(_Float16));
    }
    _Float16* hA     = (_Float16*)carve((size_t)N_NODES * HID * sizeof(_Float16));
    _Float16* hB     = (_Float16*)carve((size_t)N_NODES * HID * sizeof(_Float16));
    _Float16* aggr16 = (_Float16*)carve((size_t)N_NODES * HID * sizeof(_Float16));
    float*    sumbuf = (float*)carve(((size_t)N_NODES * HID + N_NODES) * sizeof(float));
    float*    cnt    = sumbuf + (size_t)N_NODES * HID;
    float*    hnew   = sumbuf;               // reuse: sum consumed before GEMM writes
    float*    stats  = (float*)carve(512 * sizeof(float));

    const int TB = 256;

    // fp32 -> fp16 conversions (activations input + all weights)
    cvt_f32_to_f16<<<(N_NODES * IN_DIM + TB - 1) / TB, TB, 0, stream>>>(x, xf16, N_NODES * IN_DIM);
    cvt_f32_to_f16<<<(HID * IN_DIM + TB - 1) / TB, TB, 0, stream>>>(W_emb, we16, HID * IN_DIM);
    for (int i = 0; i < 3; ++i) {
        cvt_f32_to_f16<<<(HID * HID + TB - 1) / TB, TB, 0, stream>>>(Wl[i], wl16[i], HID * HID);
        cvt_f32_to_f16<<<(HID * HID + TB - 1) / TB, TB, 0, stream>>>(Wr[i], wr16[i], HID * HID);
    }

    // embedding: h0 = x @ W_emb^T + b_emb  -> f16 hA
    wmma_gemm<<<N_NODES / 16, TB, 0, stream>>>(xf16, we16, IN_DIM,
                                               nullptr, nullptr, 0,
                                               b_emb, nullptr, hA);

    _Float16* hc = hA;
    _Float16* hn = hB;
    for (int L = 0; L < 3; ++L) {
        // zero sum + cnt (contiguous)
        zero_f32<<<(N_NODES * HID + N_NODES + TB - 1) / TB, TB, 0, stream>>>(
            sumbuf, N_NODES * HID + N_NODES);
        // fused gather + scatter-add + degree count (16 lanes/edge, 8 feats/lane)
        scatter_kernel<<<(N_EDGES * 16 + TB - 1) / TB, TB, 0, stream>>>(src, dst, hc, sumbuf, cnt);
        aggr_kernel<<<(N_NODES * HID / 8 + TB - 1) / TB, TB, 0, stream>>>(sumbuf, cnt, aggr16);

        // h_new = aggr @ Wl^T + bl + h @ Wr^T   (fp32 pre-BN, aliases sumbuf)
        wmma_gemm<<<N_NODES / 16, TB, 0, stream>>>(aggr16, wl16[L], HID,
                                                   hc, wr16[L], HID,
                                                   bl[L], hnew, nullptr);

        zero_f32<<<1, 256, 0, stream>>>(stats, 256);
        bn_stats<<<(N_NODES + 255) / 256, 128, 0, stream>>>(hnew, stats);
        bn_finalize<<<1, 128, 0, stream>>>(stats, g[L], be[L], stats + 256);
        bn_apply<<<(N_NODES * HID / 8 + TB - 1) / TB, TB, 0, stream>>>(hnew, stats + 256, hn);

        _Float16* t = hc; hc = hn; hn = t;
    }

    classifier<<<(N_NODES + TB - 1) / TB, TB, 0, stream>>>(hc, Wc, bc, (float*)d_out);
}
